// ChebyshevGCNN_1047972020814
// MI455X (gfx1250) — matrix operand order repaired and checked by reading, not compiled, not run on hardware
//
#include <hip/hip_runtime.h>

// Problem constants (match reference setup_inputs)
#define CHEB_B 4
#define CHEB_N 50000
#define CHEB_E 800000
#define CHEB_C 64   // C_IN == C_OUT
#define CHEB_K 3    // max degree -> 4 weight matrices

typedef __attribute__((ext_vector_type(16))) _Float16 v16h;
typedef __attribute__((ext_vector_type(8)))  float    v8f;

// ---------------------------------------------------------------------------
// zero a buffer (float4-vectorized)
// ---------------------------------------------------------------------------
__global__ void cheb_zero4(float4* __restrict__ p, long n4) {
    long i = (long)blockIdx.x * blockDim.x + threadIdx.x;
    if (i < n4) p[i] = make_float4(0.f, 0.f, 0.f, 0.f);
}

// ---------------------------------------------------------------------------
// Edge-parallel SpMM: dst[b, rows[e], :] += vals[e] * src[b, cols[e], :]
// one thread = one edge x 4 channels; scatter via global_atomic_add_f32.
// Per-batch node slab (12.8MB) is L2-resident on MI455X (192MB L2).
// ---------------------------------------------------------------------------
__global__ void cheb_spmm(const float* __restrict__ src, float* __restrict__ dst,
                          const int* __restrict__ rows, const int* __restrict__ cols,
                          const float* __restrict__ vals) {
    long idx = (long)blockIdx.x * blockDim.x + threadIdx.x;
    const long total = (long)CHEB_E * 16;   // 16 channel-groups of 4
    if (idx >= total) return;
    int b  = blockIdx.y;
    int e  = (int)(idx >> 4);
    int cg = ((int)idx & 15) << 2;
    float v = vals[e];
    int s = cols[e];
    int d = rows[e];
    const float4 xv = *(const float4*)(src + ((size_t)b * CHEB_N + s) * CHEB_C + cg);
    float* dp = dst + ((size_t)b * CHEB_N + d) * CHEB_C + cg;
    atomicAdd(dp + 0, v * xv.x);
    atomicAdd(dp + 1, v * xv.y);
    atomicAdd(dp + 2, v * xv.z);
    atomicAdd(dp + 3, v * xv.w);
}

// ---------------------------------------------------------------------------
// Chebyshev recurrence combine: tk = 2*tk - tkm2 (elementwise, float4)
// ---------------------------------------------------------------------------
__global__ void cheb_combine(float4* __restrict__ tk, const float4* __restrict__ tkm2,
                             long n4) {
    long i = (long)blockIdx.x * blockDim.x + threadIdx.x;
    if (i < n4) {
        float4 a = tk[i];
        float4 c = tkm2[i];
        a.x = 2.f * a.x - c.x;
        a.y = 2.f * a.y - c.y;
        a.z = 2.f * a.z - c.z;
        a.w = 2.f * a.w - c.w;
        tk[i] = a;
    }
}

// ---------------------------------------------------------------------------
// A-fragment: 16x32 f16 (M x K) per the CDNA5 VGPR layout table:
//   lane m = lane&15 picks row; lane half h = lane>>4.
//   elements 0..7  hold K = kc*32 + h*8 + 0..7
//   elements 8..15 hold K = kc*32 + 16 + h*8 + 0..7
// Rows are contiguous f32 channels in memory -> two 8-wide (2x float4) runs.
// ---------------------------------------------------------------------------
__device__ inline v16h cheb_load_a(const float* __restrict__ tileBase, int lane, int kc) {
    int m = lane & 15, h = lane >> 4;
    const float* p0 = tileBase + (size_t)m * CHEB_C + kc * 32 + h * 8;
    const float* p1 = p0 + 16;
    float4 r0a = *(const float4*)(p0);
    float4 r0b = *(const float4*)(p0 + 4);
    float4 r1a = *(const float4*)(p1);
    float4 r1b = *(const float4*)(p1 + 4);
    v16h a;
    a[0]  = (_Float16)r0a.x; a[1]  = (_Float16)r0a.y;
    a[2]  = (_Float16)r0a.z; a[3]  = (_Float16)r0a.w;
    a[4]  = (_Float16)r0b.x; a[5]  = (_Float16)r0b.y;
    a[6]  = (_Float16)r0b.z; a[7]  = (_Float16)r0b.w;
    a[8]  = (_Float16)r1a.x; a[9]  = (_Float16)r1a.y;
    a[10] = (_Float16)r1a.z; a[11] = (_Float16)r1a.w;
    a[12] = (_Float16)r1b.x; a[13] = (_Float16)r1b.y;
    a[14] = (_Float16)r1b.z; a[15] = (_Float16)r1b.w;
    return a;
}

// ---------------------------------------------------------------------------
// Fused output: out = relu(Tx0@W0 + T1@W1 + T2@W2 + T3@W3 + bias)
// 128 threads = 4 waves per block; each wave owns a 16-row x 64-col tile.
// 12500 row tiles / 4 waves = 3125 blocks exactly -> no divergence, EXEC all-1s.
//
// Weights are staged once per block into LDS PRE-SWIZZLED into the WMMA
// B-fragment register image: sFrag[frag][lane][elem], frag = k*8 + ct*2 + kc.
// B-matrix layout (32x16 KxN, f16): N = lane&15, K = (elem>>3)*16 + (lane>>4)*8 + (elem&7),
// within K-chunk kc (K += kc*32). Each fragment read is then one aligned
// 32-byte LDS load (2x ds_load_b128) instead of 16 scalar ds_load_u16.
// ---------------------------------------------------------------------------
#define CHEB_NFRAG 32   // 4 k * 4 colTiles * 2 kChunks

__global__ __launch_bounds__(128) void cheb_gemm(
        const float* __restrict__ Tx0, const float* __restrict__ T1,
        const float* __restrict__ T2,  const float* __restrict__ T3,
        const float* __restrict__ W,   const float* __restrict__ bias,
        float* __restrict__ out) {
    // 32 fragments x 32 lanes x 16 halves = 16384 f16 = 32KB
    __shared__ __align__(32) _Float16 sFrag[CHEB_NFRAG][32][16];

    // Stage weights f32 -> f16, swizzled to fragment layout; 2 elems per iter
    // (consecutive K for the same lane) packed into one b32 LDS store.
    for (int i = threadIdx.x; i < CHEB_NFRAG * 32 * 8; i += blockDim.x) {
        int frag = i >> 8;            // /256
        int r    = i & 255;
        int ln   = r >> 3;            // lane 0..31
        int ep   = (r & 7) << 1;      // elem 0,2,..,14
        int k  = frag >> 3;
        int ct = (frag >> 1) & 3;
        int kc = frag & 1;
        int n  = ln & 15, h = ln >> 4;
        int col = ct * 16 + n;
        int K0  = kc * 32 + (ep >> 3) * 16 + h * 8 + (ep & 7);
        const float* wp = W + (size_t)k * CHEB_C * CHEB_C + (size_t)K0 * CHEB_C + col;
        float w0 = wp[0];
        float w1 = wp[CHEB_C];
        union { _Float16 hx[2]; unsigned u; } pk;
        pk.hx[0] = (_Float16)w0;
        pk.hx[1] = (_Float16)w1;
        *(unsigned*)&sFrag[frag][ln][ep] = pk.u;
    }
    __syncthreads();

    int wave = threadIdx.x >> 5;
    int lane = threadIdx.x & 31;
    size_t rowBase = ((size_t)blockIdx.x * 4 + wave) * 16;   // flat row in [B*N, C]

    const float* srcs[4] = { Tx0, T1, T2, T3 };

    // Preload all A fragments: 4 cheb orders x 2 K-chunks.
    v16h a[4][2];
#pragma unroll
    for (int k = 0; k < 4; ++k)
#pragma unroll
        for (int kc = 0; kc < 2; ++kc)
            a[k][kc] = cheb_load_a(srcs[k] + rowBase * CHEB_C, lane, kc);

    int n = lane & 15, h = lane >> 4;
#pragma unroll
    for (int ct = 0; ct < 4; ++ct) {
        v8f acc = {};
#pragma unroll
        for (int k = 0; k < 4; ++k) {
#pragma unroll
            for (int kc = 0; kc < 2; ++kc) {
                // One aligned 32B LDS read per fragment.
                v16h bfr = *(const v16h*)&sFrag[k * 8 + ct * 2 + kc][lane][0];
                acc = __builtin_amdgcn_wmma_f32_16x16x32_f16(
                        /*neg_a=*/false, a[k][kc], /*neg_b=*/false, bfr,
                        /*c_mod=*/(short)0, acc, /*reuse_a=*/false, /*reuse_b=*/false);
            }
        }
        // D layout: element r -> row M = r + 8*h, col N = lane&15 (within tile ct)
        float bv = bias[ct * 16 + n];
#pragma unroll
        for (int r = 0; r < 8; ++r) {
            size_t row = rowBase + r + 8 * h;
            float v = acc[r] + bv;
            out[row * CHEB_C + ct * 16 + n] = v > 0.f ? v : 0.f;
        }
    }
}

// ---------------------------------------------------------------------------
// Host-side orchestration (all on `stream`, graph-capture safe)
// ---------------------------------------------------------------------------
extern "C" void kernel_launch(void* const* d_in, const int* in_sizes, int n_in,
                              void* d_out, int out_size, void* d_ws, size_t ws_size,
                              hipStream_t stream) {
    const float* x    = (const float*)d_in[0];
    const int*   lr   = (const int*)  d_in[1];
    const int*   lc   = (const int*)  d_in[2];
    const float* lv   = (const float*)d_in[3];
    const float* W    = (const float*)d_in[4];
    const float* bias = (const float*)d_in[5];
    float*       out  = (float*)d_out;

    const size_t bufElems = (size_t)CHEB_B * CHEB_N * CHEB_C;   // 12.8M floats
    float* T1 = (float*)d_ws;
    float* T2 = T1 + bufElems;
    float* T3 = T2 + bufElems;

    const long n4 = (long)(bufElems / 4);
    dim3 eblk(256), egrd((unsigned)((n4 + 255) / 256));                   // elementwise
    dim3 sblk(256), sgrd((unsigned)(((long)CHEB_E * 16 + 255) / 256), CHEB_B);  // spmm

    // T1 = L @ x
    cheb_zero4<<<egrd, eblk, 0, stream>>>((float4*)T1, n4);
    cheb_spmm<<<sgrd, sblk, 0, stream>>>(x, T1, lr, lc, lv);

    // T2 = 2*(L @ T1) - x
    cheb_zero4<<<egrd, eblk, 0, stream>>>((float4*)T2, n4);
    cheb_spmm<<<sgrd, sblk, 0, stream>>>(T1, T2, lr, lc, lv);
    cheb_combine<<<egrd, eblk, 0, stream>>>((float4*)T2, (const float4*)x, n4);

    // T3 = 2*(L @ T2) - T1
    cheb_zero4<<<egrd, eblk, 0, stream>>>((float4*)T3, n4);
    cheb_spmm<<<sgrd, sblk, 0, stream>>>(T2, T3, lr, lc, lv);
    cheb_combine<<<egrd, eblk, 0, stream>>>((float4*)T3, (const float4*)T1, n4);

    // out = relu(x@W0 + T1@W1 + T2@W2 + T3@W3 + bias)  via WMMA
    const int rowTiles = (CHEB_B * CHEB_N) / 16;   // 12500
    cheb_gemm<<<rowTiles / 4, 128, 0, stream>>>(x, T1, T2, T3, W, bias, out);
}
